// FlashCausalSelfAttention_44452911514404
// MI455X (gfx1250) — matrix operand order repaired and checked by reading, not compiled, not run on hardware
//
#include <hip/hip_runtime.h>
#include <hip/hip_bf16.h>
#include <math.h>

typedef __attribute__((ext_vector_type(16))) __bf16 v16bf;
typedef __attribute__((ext_vector_type(8)))  float  v8f;

// ---------------- problem constants ----------------
#define BB   2
#define TT   2048
#define CC   1024
#define NH   16
#define NKV  4
#define HD   64
#define WIN  512
#define CKV  (NKV * HD)          // 256
#define NQKV (CC + 2 * CKV)      // 1536
#define ROWS (BB * TT)           // 4096

// ---------------- fragment helpers ----------------
// A-matrix 16x32 bf16 fragment (lane = M row; K split per lane-half)
__device__ __forceinline__ v16bf load_fragA(const __bf16* base, int ld, int lane) {
  int r  = lane & 15;
  int kb = (lane >> 4) << 3;            // 0 or 8
  const __bf16* p = base + (size_t)r * ld + kb;
  v16bf f;
  float4* fp = reinterpret_cast<float4*>(&f);
  fp[0] = *reinterpret_cast<const float4*>(p);        // K = kb .. kb+7
  fp[1] = *reinterpret_cast<const float4*>(p + 16);   // K = kb+16 .. kb+23
  return f;
}

// B-matrix 32x16 bf16 fragment (lane = N column; K split per lane-half like A)
__device__ __forceinline__ v16bf load_fragB(const __bf16* base, int ld, int lane) {
  int n  = lane & 15;
  int kb = (lane >> 4) << 3;            // 0 or 8
  const __bf16* p = base + n;
  v16bf f;
#pragma unroll
  for (int s = 0; s < 8; ++s) {
    f[s]     = p[(size_t)(kb + s) * ld];
    f[8 + s] = p[(size_t)(kb + 16 + s) * ld];
  }
  return f;
}

__device__ __forceinline__ v8f wmma_bf16(v16bf a, v16bf b, v8f c) {
  return __builtin_amdgcn_wmma_f32_16x16x32_bf16(false, a, false, b, (short)0, c,
                                                 false, false);
}

// ---------------- kernel 1: convert inputs to bf16 (pack Wq|Wk|Wv) ----------------
__global__ void convert_pack(const float* __restrict__ x,
                             const float* __restrict__ wq,
                             const float* __restrict__ wk,
                             const float* __restrict__ wv,
                             const float* __restrict__ wo,
                             __bf16* __restrict__ xb,
                             __bf16* __restrict__ wqkvb,
                             __bf16* __restrict__ wob) {
  const long XCNT = (long)ROWS * CC;
  const long QCNT = (long)CC * CC;
  const long KCNT = (long)CC * CKV;
  const long TOT  = XCNT + QCNT + 2 * KCNT + QCNT;
  for (long i = blockIdx.x * (long)blockDim.x + threadIdx.x; i < TOT;
       i += (long)gridDim.x * blockDim.x) {
    long j = i;
    if (j < XCNT) { xb[j] = (__bf16)x[j]; continue; }
    j -= XCNT;
    if (j < QCNT) {  // Wq -> cols [0,1024)
      long k = j / CC, n = j % CC;
      wqkvb[k * NQKV + n] = (__bf16)wq[j];
      continue;
    }
    j -= QCNT;
    if (j < KCNT) {  // Wk -> cols [1024,1280)
      long k = j / CKV, n = j % CKV;
      wqkvb[k * NQKV + CC + n] = (__bf16)wk[j];
      continue;
    }
    j -= KCNT;
    if (j < KCNT) {  // Wv -> cols [1280,1536)
      long k = j / CKV, n = j % CKV;
      wqkvb[k * NQKV + CC + CKV + n] = (__bf16)wv[j];
      continue;
    }
    j -= KCNT;
    wob[j] = (__bf16)wo[j];
  }
}

// ---------------- kernel 2: WMMA GEMM, C(f32) = A(bf16) x B(bf16) ----------------
// one wave computes a 32x64 tile of C: 2 A-fragments reused over 4 B-fragments
// per 32-deep k-chunk -> 8 independent WMMAs per iteration, ~22 FLOP/B from L2.
__global__ void gemm_bf16_wmma(const __bf16* __restrict__ A,
                               const __bf16* __restrict__ B,
                               float* __restrict__ C, int M, int N, int K) {
  int lane = threadIdx.x & 31;
  int tile = blockIdx.x * (blockDim.x >> 5) + (threadIdx.x >> 5);
  int ntn = N >> 6;                       // 64-wide column tiles
  int ntm = M >> 5;                       // 32-high row tiles
  if (tile >= ntm * ntn) return;
  int tm = tile / ntn, tn = tile % ntn;
  int row0 = tm << 5, col0 = tn << 6;

  v8f acc[2][4];
#pragma unroll
  for (int r = 0; r < 2; ++r)
#pragma unroll
    for (int j = 0; j < 4; ++j)
      acc[r][j] = (v8f){0,0,0,0,0,0,0,0};

  const __bf16* arow0 = A + (size_t)row0 * K;
  const __bf16* arow1 = A + (size_t)(row0 + 16) * K;
  for (int k0 = 0; k0 < K; k0 += 32) {
    if (k0 + 64 < K) {
      __builtin_prefetch(arow0 + k0 + 64, 0, 1);      // global_prefetch_b8
      __builtin_prefetch(arow1 + k0 + 64, 0, 1);
    }
    v16bf a0 = load_fragA(arow0 + k0, K, lane);
    v16bf a1 = load_fragA(arow1 + k0, K, lane);
    const __bf16* bchunk = B + (size_t)k0 * N + col0;
#pragma unroll
    for (int j = 0; j < 4; ++j) {
      v16bf b = load_fragB(bchunk + j * 16, N, lane);
      acc[0][j] = wmma_bf16(a0, b, acc[0][j]);
      acc[1][j] = wmma_bf16(a1, b, acc[1][j]);
    }
  }

  int n  = lane & 15;
  int mb = (lane >> 4) << 3;
#pragma unroll
  for (int r = 0; r < 2; ++r) {
#pragma unroll
    for (int i = 0; i < 8; ++i) {
      float* crow = C + (size_t)(row0 + r * 16 + mb + i) * N + col0 + n;
      crow[0]  = acc[r][0][i];
      crow[16] = acc[r][1][i];
      crow[32] = acc[r][2][i];
      crow[48] = acc[r][3][i];
    }
  }
}

// ---------------- kernel 3: RoPE + split + relayout to bf16 ----------------
// qkv_f32: [ROWS][1536]. Outputs:
//   qb  [B][NH][T][64]  (row-major rows of Q per head -> A fragments)
//   ktb [B][NKV][64][T] (K transposed -> B fragments for S = Q*K^T)
//   vb  [B][NKV][T][64] (row-major -> B fragments for P*V)
__global__ void rope_split(const float* __restrict__ qkv,
                           __bf16* __restrict__ qb,
                           __bf16* __restrict__ ktb,
                           __bf16* __restrict__ vb) {
  long idx = blockIdx.x * (long)blockDim.x + threadIdx.x;
  const long TOT = (long)ROWS * NQKV;
  if (idx >= TOT) return;
  int col = (int)(idx % NQKV);
  long row = idx / NQKV;
  int t = (int)(row % TT);
  int b = (int)(row / TT);
  float xv = qkv[idx];

  if (col < CC + CKV) {  // q or k: apply RoPE
    int d = col & 63;
    float invf = __powf(10000.0f, -(float)(d & 31) * (1.0f / 32.0f));
    float ang  = (float)t * invf;
    float cs = cosf(ang), sn = sinf(ang);
    float partner = qkv[idx + ((d < 32) ? 32 : -32)];
    float rot = (d < 32) ? -partner : partner;
    float y = xv * cs + rot * sn;
    if (col < CC) {
      int h = col >> 6;
      qb[(((size_t)(b * NH + h)) * TT + t) * HD + d] = (__bf16)y;
    } else {
      int hk = (col - CC) >> 6;
      ktb[(((size_t)(b * NKV + hk)) * HD + d) * TT + t] = (__bf16)y;
    }
  } else {  // v: plain convert
    int c2 = col - CC - CKV;
    int hk = c2 >> 6, d = c2 & 63;
    vb[(((size_t)(b * NKV + hk)) * TT + t) * HD + d] = (__bf16)xv;
  }
}

// ---------------- kernel 4: sliding-window causal flash attention ----------------
// one wave handles one (b, h, 16-query tile); writes bf16 [ROWS][1024]
__global__ void attn_kernel(const __bf16* __restrict__ qb,
                            const __bf16* __restrict__ ktb,
                            const __bf16* __restrict__ vb,
                            const int* __restrict__ seqlens,
                            __bf16* __restrict__ aout) {
  __shared__ __bf16 plds[4 * 16 * 32];
  int lane = threadIdx.x & 31;
  int warp = threadIdx.x >> 5;
  int woff = warp * (16 * 32);
  int wid  = blockIdx.x * (blockDim.x >> 5) + warp;
  const int QT = TT / 16;
  int qt = wid % QT;
  int h  = (wid / QT) % NH;
  int b  = wid / (QT * NH);
  int hkv = h / (NH / NKV);
  int qs = qt << 4;
  int seqlen = seqlens[b];

  const __bf16* qbase  = qb  + (((size_t)(b * NH + h))   * TT + qs) * HD;
  const __bf16* ktbase = ktb + ((size_t)(b * NKV + hkv)) * HD * TT;
  const __bf16* vbase  = vb  + ((size_t)(b * NKV + hkv)) * TT * HD;

  v16bf aq0 = load_fragA(qbase,      HD, lane);   // dims 0..31
  v16bf aq1 = load_fragA(qbase + 32, HD, lane);   // dims 32..63

  v8f o0 = {0,0,0,0,0,0,0,0}, o1 = o0, o2 = o0, o3 = o0;
  float mrow[8], lrow[8];
#pragma unroll
  for (int i = 0; i < 8; ++i) { mrow[i] = -1e30f; lrow[i] = 0.0f; }

  int ncol = lane & 15;
  int mb   = (lane >> 4) << 3;
  const float SCALE = 0.125f;  // 1/sqrt(64)

  int lo = qs - WIN; if (lo < 0) lo = 0;
  int hi = qs + 15;  if (hi > seqlen - 1) hi = seqlen - 1;

  for (int ks = lo & ~31; ks <= hi; ks += 32) {
    // ---- S = Q * K^T for 32 keys (two 16-wide n-tiles) ----
    v16bf bk00 = load_fragB(ktbase + ks,                      TT, lane);
    v16bf bk10 = load_fragB(ktbase + (size_t)32 * TT + ks,    TT, lane);
    v16bf bk01 = load_fragB(ktbase + ks + 16,                 TT, lane);
    v16bf bk11 = load_fragB(ktbase + (size_t)32 * TT + ks+16, TT, lane);
    v8f s0 = {0,0,0,0,0,0,0,0}, s1 = s0;
    s0 = wmma_bf16(aq0, bk00, s0);
    s0 = wmma_bf16(aq1, bk10, s0);
    s1 = wmma_bf16(aq0, bk01, s1);
    s1 = wmma_bf16(aq1, bk11, s1);

    // ---- masked online softmax ----
#pragma unroll
    for (int i = 0; i < 8; ++i) {
      int qg  = qs + mb + i;
      int kg0 = ks + ncol;
      int kg1 = ks + 16 + ncol;
      float x0 = s0[i] * SCALE;
      float x1 = s1[i] * SCALE;
      bool v0 = (kg0 <= qg) && (kg0 >= qg - WIN) && (kg0 < seqlen);
      bool v1 = (kg1 <= qg) && (kg1 >= qg - WIN) && (kg1 < seqlen);
      x0 = v0 ? x0 : -1e30f;
      x1 = v1 ? x1 : -1e30f;
      float tmax = fmaxf(x0, x1);
#pragma unroll
      for (int m = 1; m < 16; m <<= 1) tmax = fmaxf(tmax, __shfl_xor(tmax, m, 32));
      float newm = fmaxf(mrow[i], tmax);
      float corr = __expf(mrow[i] - newm);
      float p0 = __expf(x0 - newm);
      float p1 = __expf(x1 - newm);
      float ps = p0 + p1;
#pragma unroll
      for (int m = 1; m < 16; m <<= 1) ps += __shfl_xor(ps, m, 32);
      lrow[i] = lrow[i] * corr + ps;
      mrow[i] = newm;
      o0[i] *= corr; o1[i] *= corr; o2[i] *= corr; o3[i] *= corr;
      plds[woff + (mb + i) * 32 + ncol]      = (__bf16)p0;
      plds[woff + (mb + i) * 32 + 16 + ncol] = (__bf16)p1;
    }
    asm volatile("s_wait_dscnt 0" ::: "memory");

    // ---- P (bf16, A layout) from LDS ----
    {
      int r  = lane & 15;
      int kb = (lane >> 4) << 3;
      v16bf ap;
      float4* fp = reinterpret_cast<float4*>(&ap);
      fp[0] = *reinterpret_cast<const float4*>(&plds[woff + r * 32 + kb]);
      fp[1] = *reinterpret_cast<const float4*>(&plds[woff + r * 32 + kb + 16]);

      // ---- O += P * V ----
      const __bf16* vchunk = vbase + (size_t)ks * HD;
      v16bf bv0 = load_fragB(vchunk,      HD, lane);
      v16bf bv1 = load_fragB(vchunk + 16, HD, lane);
      v16bf bv2 = load_fragB(vchunk + 32, HD, lane);
      v16bf bv3 = load_fragB(vchunk + 48, HD, lane);
      o0 = wmma_bf16(ap, bv0, o0);
      o1 = wmma_bf16(ap, bv1, o1);
      o2 = wmma_bf16(ap, bv2, o2);
      o3 = wmma_bf16(ap, bv3, o3);
    }
    asm volatile("s_wait_dscnt 0" ::: "memory");
  }

  // ---- normalize, zero padded query rows, store bf16 [row][h*64 + d] ----
#pragma unroll
  for (int i = 0; i < 8; ++i) {
    int qg = qs + mb + i;
    float fac = (qg < seqlen) ? (1.0f / lrow[i]) : 0.0f;
    __bf16* orow = aout + ((size_t)(b * TT) + qg) * CC + h * HD + ncol;
    orow[0]  = (__bf16)(o0[i] * fac);
    orow[16] = (__bf16)(o1[i] * fac);
    orow[32] = (__bf16)(o2[i] * fac);
    orow[48] = (__bf16)(o3[i] * fac);
  }
}

// ---------------- launcher ----------------
extern "C" void kernel_launch(void* const* d_in, const int* in_sizes, int n_in,
                              void* d_out, int out_size, void* d_ws, size_t ws_size,
                              hipStream_t stream) {
  const float* x  = (const float*)d_in[0];
  const int* seq  = (const int*)d_in[1];
  const float* wq = (const float*)d_in[2];
  const float* wk = (const float*)d_in[3];
  const float* wv = (const float*)d_in[4];
  const float* wo = (const float*)d_in[5];
  float* out = (float*)d_out;

  char* ws = (char*)d_ws;
  size_t off = 0;
  __bf16* xb    = (__bf16*)(ws + off); off += (size_t)ROWS * CC * 2;          // 8 MiB
  __bf16* wqkvb = (__bf16*)(ws + off); off += (size_t)CC * NQKV * 2;          // 3 MiB
  __bf16* wob   = (__bf16*)(ws + off); off += (size_t)CC * CC * 2;            // 2 MiB
  float*  qkvf  = (float*) (ws + off); off += (size_t)ROWS * NQKV * 4;        // 24 MiB
  __bf16* qb    = (__bf16*)(ws + off); off += (size_t)BB * NH * TT * HD * 2;  // 8 MiB
  __bf16* ktb   = (__bf16*)(ws + off); off += (size_t)BB * NKV * HD * TT * 2; // 2 MiB
  __bf16* vb    = (__bf16*)(ws + off); off += (size_t)BB * NKV * TT * HD * 2; // 2 MiB
  __bf16* ab    = (__bf16*)(ws + off); off += (size_t)ROWS * CC * 2;          // 8 MiB

  // 1) convert inputs to bf16 (pack Wq|Wk|Wv)
  convert_pack<<<4096, 256, 0, stream>>>(x, wq, wk, wv, wo, xb, wqkvb, wob);

  // 2) fused QKV projection: [4096x1024] x [1024x1536], 32x64 tiles
  {
    int tiles = (ROWS / 32) * (NQKV / 64);   // 3072 waves
    gemm_bf16_wmma<<<tiles / 4, 128, 0, stream>>>(xb, wqkvb, qkvf, ROWS, NQKV, CC);
  }

  // 3) RoPE + split + relayout
  {
    long tot = (long)ROWS * NQKV;
    rope_split<<<(int)((tot + 255) / 256), 256, 0, stream>>>(qkvf, qb, ktb, vb);
  }

  // 4) flash attention (one wave per 16-query tile per head)
  {
    int waves = BB * NH * (TT / 16);         // 4096 waves
    attn_kernel<<<waves / 4, 128, 0, stream>>>(qb, ktb, vb, seq, ab);
  }

  // 5) output projection: [4096x1024] x [1024x1024] -> f32 d_out, 32x64 tiles
  {
    int tiles = (ROWS / 32) * (CC / 64);     // 2048 waves
    gemm_bf16_wmma<<<tiles / 4, 128, 0, stream>>>(ab, wob, out, ROWS, CC, CC);
  }
}
